// BuildingGenModel_87522843557953
// MI455X (gfx1250) — matrix operand order repaired and checked by reading, not compiled, not run on hardware
//
#include <hip/hip_runtime.h>

typedef __attribute__((ext_vector_type(16))) _Float16 v16h;
typedef __attribute__((ext_vector_type(8)))  _Float16 v8h;
typedef __attribute__((ext_vector_type(8)))  float    v8f;

#define BN_EPS 1e-5f

// ---------- order-preserving float<->uint encoding for atomic min ----------
__device__ __forceinline__ unsigned enc_f32(float x) {
  unsigned u = __float_as_uint(x);
  return (u & 0x80000000u) ? ~u : (u | 0x80000000u);
}
__device__ __forceinline__ float dec_f32(unsigned e) {
  unsigned u = (e & 0x80000000u) ? (e & 0x7fffffffu) : ~e;
  return __uint_as_float(u);
}

// ---------- utility kernels ----------
__global__ void k_fill_u32(unsigned* __restrict__ p, unsigned v, long long n) {
  long long i = (long long)blockIdx.x * blockDim.x + threadIdx.x;
  if (i < n) p[i] = v;
}

__global__ void k_degree(const int* __restrict__ dst, int* __restrict__ deg, int E) {
  int e = blockIdx.x * blockDim.x + threadIdx.x;
  if (e < E) atomicAdd(&deg[dst[e]], 1);
}

// scatter-min: aggr[dst] = min(aggr[dst], h[src]) over all edges, per feature
__global__ void k_scatter_min(const float* __restrict__ h, const int* __restrict__ src,
                              const int* __restrict__ dst, unsigned* __restrict__ enc,
                              int E, int F) {
  long long i = (long long)blockIdx.x * blockDim.x + threadIdx.x;
  long long total = (long long)E * F;
  if (i >= total) return;
  int e = (int)(i / F);
  int f = (int)(i - (long long)e * F);
  float v = h[(long long)src[e] * F + f];
  atomicMin(&enc[(long long)dst[e] * F + f], enc_f32(v));
}

// decode encoded mins -> padded f16 operand; nodes with no in-edges -> 0
__global__ void k_finalize_aggr(const unsigned* __restrict__ enc, const int* __restrict__ deg,
                                _Float16* __restrict__ out, int Nn, int F, int Kp) {
  long long i = (long long)blockIdx.x * blockDim.x + threadIdx.x;
  if (i >= (long long)Nn * Kp) return;
  int n = (int)(i / Kp);
  int f = (int)(i - (long long)n * Kp);
  float v = 0.0f;
  if (f < F && deg[n] > 0) v = dec_f32(enc[(long long)n * F + f]);
  out[i] = (_Float16)v;
}

// fp32 [Rows,F] row-major -> f16 [Rows,Kp] zero padded (activations and weights)
__global__ void k_convert_f16(const float* __restrict__ src, _Float16* __restrict__ out,
                              int Rows, int F, int Kp) {
  long long i = (long long)blockIdx.x * blockDim.x + threadIdx.x;
  if (i >= (long long)Rows * Kp) return;
  int r = (int)(i / Kp);
  int f = (int)(i - (long long)r * Kp);
  out[i] = (f < F) ? (_Float16)src[(long long)r * F + f] : (_Float16)0.0f;
}

// ---------- WMMA fused SAGE GEMM: out = A1 @ W1^T + A2 @ W2^T + bias ----------
// A1/A2: [Nn, Kp] f16 row-major;  W1/W2: [Fout, Kp] f16 row-major (so B = W^T)
// One wave computes a 16x16 output tile; K-loop in chunks of 32.
__global__ void k_sage_gemm(const _Float16* __restrict__ A1, const _Float16* __restrict__ A2,
                            const _Float16* __restrict__ W1, const _Float16* __restrict__ W2,
                            const float* __restrict__ bias, float* __restrict__ out,
                            int Nn, int Kp, int Fout) {
  const int wave  = threadIdx.x >> 5;
  const int lane  = threadIdx.x & 31;
  const int mtile = blockIdx.x * (blockDim.x >> 5) + wave;
  const int ntiles = Nn >> 4;              // Nn multiple of 16 (50000 = 3125*16)
  if (mtile >= ntiles) return;             // wave-uniform: EXEC stays all-1 for WMMA
  const int m    = lane & 15;
  const int half = lane >> 4;
  const int row0 = mtile << 4;
  const int col0 = blockIdx.y << 4;

  v8f acc = {};
  const int nchunks = Kp >> 5;
  #pragma unroll 1
  for (int pass = 0; pass < 2; ++pass) {
    const _Float16* A = pass ? A2 : A1;
    const _Float16* W = pass ? W2 : W1;
    const _Float16* arow = A + (long long)(row0 + m) * Kp;
    const _Float16* wrow = W + (long long)(col0 + m) * Kp;
    for (int c = 0; c < nchunks; ++c) {
      const int k0 = c << 5;
      // A fragment (ISA 16-bit A 16x32 layout): lane holds K = 8*half+(0..7) and 16+8*half+(0..7)
      v8h alo = *(const v8h*)(arow + k0 + half * 8);
      v8h ahi = *(const v8h*)(arow + k0 + 16 + half * 8);
      v16h a;
      #pragma unroll
      for (int j = 0; j < 8; ++j) { a[j] = alo[j]; a[j + 8] = ahi[j]; }
      // B fragment: lane n=m holds K = 8*half+(0..7) and 16+8*half+(0..7) of column n of B (= row n of W)
      v8h blo = *(const v8h*)(wrow + k0 + half * 8);
      v8h bhi = *(const v8h*)(wrow + k0 + 16 + half * 8);
      v16h b;
      #pragma unroll
      for (int j = 0; j < 8; ++j) { b[j] = blo[j]; b[j + 8] = bhi[j]; }
      acc = __builtin_amdgcn_wmma_f32_16x16x32_f16(false, a, false, b,
                                                   (short)0, acc, false, false);
    }
  }
  const float bv = bias[col0 + m];
  #pragma unroll
  for (int r = 0; r < 8; ++r)
    out[(long long)(row0 + half * 8 + r) * Fout + col0 + m] = acc[r] + bv;
}

// ---------- BatchNorm (training-mode batch stats, deterministic reduction) ----------
__global__ void k_bn_stats(const float* __restrict__ h, float* __restrict__ mean,
                           float* __restrict__ rstd, int Nn, int F) {
  __shared__ float ss[256];
  __shared__ float sq[256];
  const int f = blockIdx.x;
  const int t = threadIdx.x;
  float s = 0.f, q = 0.f;
  for (int n = t; n < Nn; n += 256) {
    float v = h[(long long)n * F + f];
    s += v; q += v * v;
  }
  ss[t] = s; sq[t] = q;
  __syncthreads();
  for (int w = 128; w > 0; w >>= 1) {
    if (t < w) { ss[t] += ss[t + w]; sq[t] += sq[t + w]; }
    __syncthreads();
  }
  if (t == 0) {
    float m = ss[0] / (float)Nn;
    float v = sq[0] / (float)Nn - m * m;      // biased variance (jnp.var)
    mean[f] = m;
    rstd[f] = rsqrtf(v + BN_EPS);
  }
}

__global__ void k_bn_relu(float* __restrict__ h, const float* __restrict__ mean,
                          const float* __restrict__ rstd, const float* __restrict__ g,
                          const float* __restrict__ b, _Float16* __restrict__ h16,
                          int Nn, int F) {
  long long i = (long long)blockIdx.x * blockDim.x + threadIdx.x;
  if (i >= (long long)Nn * F) return;
  int f = (int)(i % F);
  float v = (h[i] - mean[f]) * rstd[f] * g[f] + b[f];
  v = fmaxf(v, 0.0f);
  h[i] = v;
  if (h16) h16[i] = (_Float16)v;              // Kp == F for these layers
}

// ---------- small layers ----------
// rm4: [Nn,3] logits = aggr(f16,Kp=128)@Wl^T + bl + h(f32)@Wr^T
__global__ void k_rm4(const _Float16* __restrict__ aggr16, const float* __restrict__ h,
                      const float* __restrict__ Wl, const float* __restrict__ bl,
                      const float* __restrict__ Wr, float* __restrict__ logits, int Nn) {
  int n = blockIdx.x * blockDim.x + threadIdx.x;
  if (n >= Nn) return;
  const _Float16* a = aggr16 + (long long)n * 128;
  const float* hv = h + (long long)n * 128;
  #pragma unroll
  for (int o = 0; o < 3; ++o) {
    const float* wl = Wl + o * 128;
    const float* wr = Wr + o * 128;
    float s = bl[o];
    for (int k = 0; k < 128; ++k) s += (float)a[k] * wl[k] + hv[k] * wr[k];
    logits[(long long)n * 3 + o] = s;
  }
}

__global__ void k_lsm_mask(const float* __restrict__ logits, float* __restrict__ out_ls,
                           float* __restrict__ mask, int Nn) {
  int n = blockIdx.x * blockDim.x + threadIdx.x;
  if (n >= Nn) return;
  float v0 = logits[n * 3 + 0], v1 = logits[n * 3 + 1], v2 = logits[n * 3 + 2];
  int best = 0; float bv = v0;
  if (v1 > bv) { bv = v1; best = 1; }
  if (v2 > bv) { bv = v2; best = 2; }
  float mx = fmaxf(v0, fmaxf(v1, v2));
  float lse = mx + logf(expf(v0 - mx) + expf(v1 - mx) + expf(v2 - mx));
  out_ls[n * 3 + 0] = v0 - lse;
  out_ls[n * 3 + 1] = v1 - lse;
  out_ls[n * 3 + 2] = v2 - lse;
  mask[n] = (best == 2) ? 1.0f : 0.0f;
}

__global__ void k_concat(const float* __restrict__ x, const float* __restrict__ logits,
                         float* __restrict__ xcat, int Nn) {
  long long i = (long long)blockIdx.x * blockDim.x + threadIdx.x;
  if (i >= (long long)Nn * 35) return;
  int n = (int)(i / 35);
  int f = (int)(i - (long long)n * 35);
  xcat[i] = (f < 32) ? x[(long long)n * 32 + f] : logits[(long long)n * 3 + (f - 32)];
}

// head: out[n] = (h[n,:128] . W + b) * mask[n]
__global__ void k_head(const float* __restrict__ h, const float* __restrict__ W,
                       const float* __restrict__ b, const float* __restrict__ mask,
                       float* __restrict__ out, int Nn) {
  int n = blockIdx.x * blockDim.x + threadIdx.x;
  if (n >= Nn) return;
  const float* hv = h + (long long)n * 128;
  float s = b[0];
  for (int k = 0; k < 128; ++k) s += hv[k] * W[k];
  out[n] = s * mask[n];
}

// ---------- host-side helpers ----------
static inline dim3 grid1(long long n, int bs) { return dim3((unsigned)((n + bs - 1) / bs)); }

static void run_gemm(const _Float16* A1, const _Float16* A2, const _Float16* W1,
                     const _Float16* W2, const float* bias, float* out,
                     int Nn, int Kp, int Fout, hipStream_t s) {
  dim3 grid((Nn / 16 + 3) / 4, Fout / 16);
  k_sage_gemm<<<grid, dim3(128), 0, s>>>(A1, A2, W1, W2, bias, out, Nn, Kp, Fout);
}

static void run_bn(float* h, const float* g, const float* b, float* mean, float* rstd,
                   _Float16* h16, int Nn, int F, hipStream_t s) {
  k_bn_stats<<<dim3(F), dim3(256), 0, s>>>(h, mean, rstd, Nn, F);
  k_bn_relu<<<grid1((long long)Nn * F, 256), dim3(256), 0, s>>>(h, mean, rstd, g, b, h16, Nn, F);
}

static void run_aggr(const float* h, const int* srcE, const int* dstE, const int* deg,
                     unsigned* enc, _Float16* aggr16, int Nn, int E, int F, int Kp,
                     hipStream_t s) {
  k_fill_u32<<<grid1((long long)Nn * F, 256), dim3(256), 0, s>>>(enc, 0xFFFFFFFFu, (long long)Nn * F);
  k_scatter_min<<<grid1((long long)E * F, 256), dim3(256), 0, s>>>(h, srcE, dstE, enc, E, F);
  k_finalize_aggr<<<grid1((long long)Nn * Kp, 256), dim3(256), 0, s>>>(enc, deg, aggr16, Nn, F, Kp);
}

extern "C" void kernel_launch(void* const* d_in, const int* in_sizes, int n_in,
                              void* d_out, int out_size, void* d_ws, size_t ws_size,
                              hipStream_t stream) {
  // ---- inputs (setup_inputs dict order) ----
  const float* x       = (const float*)d_in[0];
  const int*   edges   = (const int*)d_in[1];
  const float* rm1_Wl  = (const float*)d_in[2];
  const float* rm1_bl  = (const float*)d_in[3];
  const float* rm1_Wr  = (const float*)d_in[4];
  const float* rmn1_g  = (const float*)d_in[5];
  const float* rmn1_b  = (const float*)d_in[6];
  const float* rm2_Wl  = (const float*)d_in[7];
  const float* rm2_bl  = (const float*)d_in[8];
  const float* rm2_Wr  = (const float*)d_in[9];
  const float* rmn2_g  = (const float*)d_in[10];
  const float* rmn2_b  = (const float*)d_in[11];
  const float* rm4_Wl  = (const float*)d_in[12];
  const float* rm4_bl  = (const float*)d_in[13];
  const float* rm4_Wr  = (const float*)d_in[14];
  const float* sh1_Wl  = (const float*)d_in[15];
  const float* sh1_bl  = (const float*)d_in[16];
  const float* sh1_Wr  = (const float*)d_in[17];
  const float* shn1_g  = (const float*)d_in[18];
  const float* shn1_b  = (const float*)d_in[19];
  const float* pm1_Wl  = (const float*)d_in[20];
  const float* pm1_bl  = (const float*)d_in[21];
  const float* pm1_Wr  = (const float*)d_in[22];
  const float* pmn1_g  = (const float*)d_in[23];
  const float* pmn1_b  = (const float*)d_in[24];
  const float* pm3_W   = (const float*)d_in[25];
  const float* pm3_b   = (const float*)d_in[26];
  const float* sm1_Wl  = (const float*)d_in[27];
  const float* sm1_bl  = (const float*)d_in[28];
  const float* sm1_Wr  = (const float*)d_in[29];
  const float* smn1_g  = (const float*)d_in[30];
  const float* smn1_b  = (const float*)d_in[31];
  const float* sm3_W   = (const float*)d_in[32];
  const float* sm3_b   = (const float*)d_in[33];

  const int Nn = in_sizes[0] / 32;     // 50000, multiple of 16
  const int E  = in_sizes[1] / 2;      // 800000
  const int* srcE = edges;
  const int* dstE = edges + E;
  float* out = (float*)d_out;          // [3N | N | N]

  // ---- workspace carving ----
  size_t off = 0;
  auto wsalloc = [&](size_t bytes) -> void* {
    void* p = (char*)d_ws + off;
    off += (bytes + 255) & ~(size_t)255;
    return p;
  };
  int*       deg     = (int*)      wsalloc((size_t)Nn * 4);
  unsigned*  enc     = (unsigned*) wsalloc((size_t)Nn * 128 * 4);
  _Float16*  aggr16  = (_Float16*) wsalloc((size_t)Nn * 128 * 2);
  _Float16*  root16  = (_Float16*) wsalloc((size_t)Nn * 128 * 2);
  float*     act1    = (float*)    wsalloc((size_t)Nn * 128 * 4);
  float*     act2    = (float*)    wsalloc((size_t)Nn * 128 * 4);
  float*     xs      = (float*)    wsalloc((size_t)Nn * 128 * 4);
  float*     xcat    = (float*)    wsalloc((size_t)Nn * 35 * 4);
  float*     logits  = (float*)    wsalloc((size_t)Nn * 3 * 4);
  float*     mask    = (float*)    wsalloc((size_t)Nn * 4);
  float*     mean    = (float*)    wsalloc(128 * 4);
  float*     rstd    = (float*)    wsalloc(128 * 4);
  _Float16*  w_rm1l  = (_Float16*) wsalloc((size_t)128 * 32 * 2);
  _Float16*  w_rm1r  = (_Float16*) wsalloc((size_t)128 * 32 * 2);
  _Float16*  w_rm2l  = (_Float16*) wsalloc((size_t)128 * 128 * 2);
  _Float16*  w_rm2r  = (_Float16*) wsalloc((size_t)128 * 128 * 2);
  _Float16*  w_sh1l  = (_Float16*) wsalloc((size_t)128 * 64 * 2);
  _Float16*  w_sh1r  = (_Float16*) wsalloc((size_t)128 * 64 * 2);
  _Float16*  w_pm1l  = (_Float16*) wsalloc((size_t)128 * 128 * 2);
  _Float16*  w_pm1r  = (_Float16*) wsalloc((size_t)128 * 128 * 2);
  _Float16*  w_sm1l  = (_Float16*) wsalloc((size_t)128 * 128 * 2);
  _Float16*  w_sm1r  = (_Float16*) wsalloc((size_t)128 * 128 * 2);
  (void)ws_size; (void)n_in; (void)out_size;

  // ---- weight conversion (fp32 -> padded f16) ----
  k_convert_f16<<<grid1(128 * 32, 256),  dim3(256), 0, stream>>>(rm1_Wl, w_rm1l, 128, 32, 32);
  k_convert_f16<<<grid1(128 * 32, 256),  dim3(256), 0, stream>>>(rm1_Wr, w_rm1r, 128, 32, 32);
  k_convert_f16<<<grid1(128 * 128, 256), dim3(256), 0, stream>>>(rm2_Wl, w_rm2l, 128, 128, 128);
  k_convert_f16<<<grid1(128 * 128, 256), dim3(256), 0, stream>>>(rm2_Wr, w_rm2r, 128, 128, 128);
  k_convert_f16<<<grid1(128 * 64, 256),  dim3(256), 0, stream>>>(sh1_Wl, w_sh1l, 128, 35, 64);
  k_convert_f16<<<grid1(128 * 64, 256),  dim3(256), 0, stream>>>(sh1_Wr, w_sh1r, 128, 35, 64);
  k_convert_f16<<<grid1(128 * 128, 256), dim3(256), 0, stream>>>(pm1_Wl, w_pm1l, 128, 128, 128);
  k_convert_f16<<<grid1(128 * 128, 256), dim3(256), 0, stream>>>(pm1_Wr, w_pm1r, 128, 128, 128);
  k_convert_f16<<<grid1(128 * 128, 256), dim3(256), 0, stream>>>(sm1_Wl, w_sm1l, 128, 128, 128);
  k_convert_f16<<<grid1(128 * 128, 256), dim3(256), 0, stream>>>(sm1_Wr, w_sm1r, 128, 128, 128);

  // ---- degree (has_in) ----
  k_fill_u32<<<grid1(Nn, 256), dim3(256), 0, stream>>>((unsigned*)deg, 0u, Nn);
  k_degree<<<grid1(E, 256), dim3(256), 0, stream>>>(dstE, deg, E);

  // ---- rm1: x(32) -> act1(128) ----
  run_aggr(x, srcE, dstE, deg, enc, aggr16, Nn, E, 32, 32, stream);
  k_convert_f16<<<grid1((long long)Nn * 32, 256), dim3(256), 0, stream>>>(x, root16, Nn, 32, 32);
  run_gemm(aggr16, root16, w_rm1l, w_rm1r, rm1_bl, act1, Nn, 32, 128, stream);
  run_bn(act1, rmn1_g, rmn1_b, mean, rstd, root16, Nn, 128, stream);  // root16 = act1 f16

  // ---- rm2: act1(128) -> act2(128) ----
  run_aggr(act1, srcE, dstE, deg, enc, aggr16, Nn, E, 128, 128, stream);
  run_gemm(aggr16, root16, w_rm2l, w_rm2r, rm2_bl, act2, Nn, 128, 128, stream);
  run_bn(act2, rmn2_g, rmn2_b, mean, rstd, nullptr, Nn, 128, stream);

  // ---- rm4: act2(128) -> logits(3) ----
  run_aggr(act2, srcE, dstE, deg, enc, aggr16, Nn, E, 128, 128, stream);
  k_rm4<<<grid1(Nn, 256), dim3(256), 0, stream>>>(aggr16, act2, rm4_Wl, rm4_bl, rm4_Wr, logits, Nn);

  // ---- log_softmax + mask, concat ----
  k_lsm_mask<<<grid1(Nn, 256), dim3(256), 0, stream>>>(logits, out, mask, Nn);
  k_concat<<<grid1((long long)Nn * 35, 256), dim3(256), 0, stream>>>(x, logits, xcat, Nn);

  // ---- sh1: xcat(35 -> pad 64) -> xs(128) ----
  run_aggr(xcat, srcE, dstE, deg, enc, aggr16, Nn, E, 35, 64, stream);
  k_convert_f16<<<grid1((long long)Nn * 64, 256), dim3(256), 0, stream>>>(xcat, root16, Nn, 35, 64);
  run_gemm(aggr16, root16, w_sh1l, w_sh1r, sh1_bl, xs, Nn, 64, 128, stream);
  run_bn(xs, shn1_g, shn1_b, mean, rstd, root16, Nn, 128, stream);    // root16 = xs f16

  // ---- shared aggregation of xs for pm1 and sm1 ----
  run_aggr(xs, srcE, dstE, deg, enc, aggr16, Nn, E, 128, 128, stream);

  // ---- pm1 + pm3 (reuse act1) ----
  run_gemm(aggr16, root16, w_pm1l, w_pm1r, pm1_bl, act1, Nn, 128, 128, stream);
  run_bn(act1, pmn1_g, pmn1_b, mean, rstd, nullptr, Nn, 128, stream);
  k_head<<<grid1(Nn, 256), dim3(256), 0, stream>>>(act1, pm3_W, pm3_b, mask, out + 3LL * Nn, Nn);

  // ---- sm1 + sm3 (reuse act2) ----
  run_gemm(aggr16, root16, w_sm1l, w_sm1r, sm1_bl, act2, Nn, 128, 128, stream);
  run_bn(act2, smn1_g, smn1_b, mean, rstd, nullptr, Nn, 128, stream);
  k_head<<<grid1(Nn, 256), dim3(256), 0, stream>>>(act2, sm3_W, sm3_b, mask, out + 4LL * Nn, Nn);
}
